// RGLRU_82506321756898
// MI455X (gfx1250) — matrix operand order repaired and checked by reading, not compiled, not run on hardware
//
#include <hip/hip_runtime.h>
#include <hip/hip_bf16.h>
#include <math.h>

// ---------------- problem constants ----------------
static constexpr int Bc   = 4;
static constexpr int Sc   = 4096;
static constexpr int DIN  = 1024;
static constexpr int Hc   = 1024;
static constexpr int DOUT = 1024;
static constexpr int PE   = 16;
static constexpr int Mtot = Bc * Sc;          // 16384 rows (b,s)
static constexpr int KF   = DIN + PE;          // 1040
static constexpr int KFP  = 1088;              // padded to multiple of 64 (17*64)

typedef __attribute__((ext_vector_type(16))) __bf16 v16bf;
typedef __attribute__((ext_vector_type(8)))  __bf16 v8bf;
typedef __attribute__((ext_vector_type(8)))  float  v8f;

// LDS tile geometry: A 64x64, B 128x64, rows padded to 72 bf16 (144B = 36 dwords;
// 36*l mod 64 distinct for l=0..15 -> conflict-free 16-lane b128 reads, 16B aligned)
static constexpr int KSTEP   = 64;
static constexpr int LSTR    = 72;                 // padded row stride (bf16 elems)
static constexpr int A_ROWS  = 64;
static constexpr int B_ROWS  = 128;
static constexpr int BUF_ELE = (A_ROWS + B_ROWS) * LSTR;   // elems per buffer (13824)

// ---------------- prep: build bf16 feature matrix [Mtot x KFP] ----------------
__global__ __launch_bounds__(256) void prep_x_kernel(const float* __restrict__ x,
                                                     __bf16* __restrict__ Xf, int total) {
    int i = blockIdx.x * blockDim.x + threadIdx.x;
    if (i >= total) return;
    int m = i / KFP;
    int k = i - m * KFP;
    float v = 0.f;
    if (k < DIN) {
        v = x[(size_t)m * DIN + k];
    } else if (k < KF) {
        int s = m & (Sc - 1);
        int j = k - DIN;
        int pair = j >> 1;
        float div = __expf((float)(2 * pair) * (-9.210340371976184f / (float)PE));
        float ang = (float)s * div;
        v = ((j & 1) ? __cosf(ang) : __sinf(ang)) * 0.01f;
    }
    Xf[i] = (__bf16)v;
}

// ---------------- prep: transpose+convert weight [K x N] f32 -> [N x Kpad] bf16 ----------------
__global__ __launch_bounds__(256) void prep_wt_kernel(const float* __restrict__ W,
                                                      __bf16* __restrict__ Wt,
                                                      int K, int N, int Kpad, int total) {
    int i = blockIdx.x * blockDim.x + threadIdx.x;
    if (i >= total) return;
    int n = i / Kpad;
    int k = i - n * Kpad;
    Wt[i] = (__bf16)((k < K) ? W[(size_t)k * N + n] : 0.f);
}

// ---------------- CDNA5 async global->LDS copy (ASYNCcnt-tracked) ----------------
__device__ inline void async_ld_b128(unsigned lds_byte, const void* base, unsigned gbyte) {
    // GVS mode: 64-bit SGPR base + 32-bit per-lane byte offset; LDS dest byte addr in VGPR
    asm volatile("global_load_async_to_lds_b128 %0, %1, %2"
                 :
                 : "v"(lds_byte), "v"(gbyte), "s"(base)
                 : "memory");
}

__device__ inline void wait_async0() {
    asm volatile("s_wait_asynccnt 0x0" ::: "memory");
}

// Stage one K-step tile: A 64x64 (512 x 16B chunks) + B 128x64 (1024 chunks)
// 256 threads: 2 A-chunks + 4 B-chunks each, all as async b128 copies.
__device__ inline void stage_tile(const __bf16* __restrict__ A, const __bf16* __restrict__ Wt,
                                  int lda, int ldb, int m0blk, int n0blk, int k0,
                                  int t, unsigned lds_base_byte) {
#pragma unroll
    for (int i = 0; i < 2; ++i) {   // A tile: row = idx>>3, 16B chunk = idx&7
        int idx = t * 2 + i;
        int row = idx >> 3, ch = idx & 7;
        unsigned lds = lds_base_byte + (unsigned)(row * LSTR + ch * 8) * 2u;
        unsigned g   = ((unsigned)(m0blk + row) * (unsigned)lda + (unsigned)(k0 + ch * 8)) * 2u;
        async_ld_b128(lds, A, g);
    }
#pragma unroll
    for (int i = 0; i < 4; ++i) {   // B tile
        int idx = t * 4 + i;
        int row = idx >> 3, ch = idx & 7;
        unsigned lds = lds_base_byte + (unsigned)((A_ROWS + row) * LSTR + ch * 8) * 2u;
        unsigned g   = ((unsigned)(n0blk + row) * (unsigned)ldb + (unsigned)(k0 + ch * 8)) * 2u;
        async_ld_b128(lds, Wt, g);
    }
}

// ---------------- WMMA fragment loads from LDS (wave32, 16x16x32 bf16) ----------------
__device__ inline v16bf lds_frag_A(const __bf16* Ab, int row0, int kk, int lane) {
    // lanes 0-15 -> M=lane, K 0..7 & 16..23; lanes 16-31 -> M=lane-16, K 8..15 & 24..31
    int r  = row0 + (lane & 15);
    int kh = (lane >> 4) << 3;
    const __bf16* p = Ab + r * LSTR + kk + kh;
    union { v16bf v; v8bf h[2]; } u;
    u.h[0] = *(const v8bf*)(p);
    u.h[1] = *(const v8bf*)(p + 16);
    return u.v;
}

__device__ inline v16bf lds_frag_B(const __bf16* Bb, int col0, int kk, int lane) {
    // lanes 0-15 -> N=lane, K 0..7 & 16..23; lanes 16-31 -> N=lane-16, K 8..15 & 24..31
    int n  = col0 + (lane & 15);
    int kh = (lane >> 4) << 3;
    const __bf16* p = Bb + n * LSTR + kk + kh;
    union { v16bf v; v8bf h[2]; } u;
    u.h[0] = *(const v8bf*)(p);
    u.h[1] = *(const v8bf*)(p + 16);
    return u.v;
}

// ---------------- epilogue store of one 16x16 f32 accumulator ----------------
// MODE: 0 = dx (f32 = acc+bias)
//       1 = xh (bf16 = (acc+bias)*sqrt(1-sig(a_logit[n])^2))
//       2 = xg (bf16 = sigmoid(acc+bias) * xh[m,n])          aux = xh (bf16)
//       3 = a  (bf16 = sigmoid(acc+bias) * sig(a_logit[n]))
//       4 = out(f32 = (dx[m,n] + acc + bias) * 0.5)          aux = dx (f32)
template <int MODE>
__device__ inline void store_acc(v8f acc, int mb, int nb, int lane, int N,
                                 const float* __restrict__ bias,
                                 const float* __restrict__ a_logit,
                                 const void* __restrict__ aux,
                                 float* __restrict__ of, __bf16* __restrict__ ob) {
    int n    = nb + (lane & 15);
    int mrow = mb + ((lane >> 4) << 3);
    float bn = bias[n];
    float sig_a = 0.f;
    if (MODE == 1 || MODE == 3) {
        float al = a_logit[n];
        sig_a = 1.f / (1.f + __expf(-al));
    }
#pragma unroll
    for (int v = 0; v < 8; ++v) {
        int m = mrow + v;
        size_t idx = (size_t)m * N + n;
        float val = acc[v] + bn;
        if (MODE == 0) {
            of[idx] = val;
        } else if (MODE == 1) {
            ob[idx] = (__bf16)(val * sqrtf(fmaxf(0.f, 1.f - sig_a * sig_a)));
        } else if (MODE == 2) {
            float g = 1.f / (1.f + __expf(-val));
            ob[idx] = (__bf16)(g * (float)((const __bf16*)aux)[idx]);
        } else if (MODE == 3) {
            float g = 1.f / (1.f + __expf(-val));
            ob[idx] = (__bf16)(g * sig_a);
        } else {
            of[idx] = (((const float*)aux)[idx] + val) * 0.5f;
        }
    }
}

// ---------------- async double-buffered WMMA GEMM ----------------
// C[Mtot x N] = A[Mtot x K] @ Wt[N x K]^T
// 256 threads = 8 waves; block tile 64(M) x 128(N); wave tile 32x32; K step 64
// (8 WMMAs + 16 ds_load_b128 per wave per barrier phase).
// Next tile staged into LDS via GLOBAL_LOAD_ASYNC_TO_LDS_B128 while current computes.
template <int MODE>
__global__ __launch_bounds__(256) void gemm_wmma_kernel(
    const __bf16* __restrict__ A, const __bf16* __restrict__ Wt,
    int K, int N, int lda, int ldb,
    const float* __restrict__ bias, const float* __restrict__ a_logit,
    const void* __restrict__ aux, float* __restrict__ of, __bf16* __restrict__ ob) {
    __shared__ __attribute__((aligned(128))) __bf16 smem[2 * BUF_ELE];

    int t = threadIdx.x;
    int lane = t & 31;
    int wave = t >> 5;
    int wm = wave >> 2;                   // 0..1
    int wn = wave & 3;                    // 0..3
    int m0blk = blockIdx.y * 64;
    int n0blk = blockIdx.x * 128;
    int mw = wm * 32;                     // local M offset in A tile
    int nw = wn * 32;                     // local N offset in B tile

    v8f c00 = {}, c01 = {}, c10 = {}, c11 = {};

    // prologue: stage first tile into buffer 0
    stage_tile(A, Wt, lda, ldb, m0blk, n0blk, 0, t, 0u);

    int T = K / KSTEP;
    unsigned cur = 0;
    for (int kt = 0; kt < T; ++kt) {
        wait_async0();        // this wave's async copies have landed in LDS
        __syncthreads();      // everyone's copies have landed

        if (kt + 1 < T)       // stage next tile into the other buffer
            stage_tile(A, Wt, lda, ldb, m0blk, n0blk, (kt + 1) * KSTEP, t,
                       (1u - cur) * (unsigned)(BUF_ELE * 2));

        const __bf16* Ab = smem + cur * BUF_ELE;
        const __bf16* Bb = Ab + A_ROWS * LSTR;
#pragma unroll
        for (int kk = 0; kk < KSTEP; kk += 32) {
            v16bf a0 = lds_frag_A(Ab, mw,      kk, lane);
            v16bf a1 = lds_frag_A(Ab, mw + 16, kk, lane);
            v16bf b0 = lds_frag_B(Bb, nw,      kk, lane);
            v16bf b1 = lds_frag_B(Bb, nw + 16, kk, lane);
            c00 = __builtin_amdgcn_wmma_f32_16x16x32_bf16(false, a0, false, b0, (short)0, c00, false, false);
            c01 = __builtin_amdgcn_wmma_f32_16x16x32_bf16(false, a0, false, b1, (short)0, c01, false, false);
            c10 = __builtin_amdgcn_wmma_f32_16x16x32_bf16(false, a1, false, b0, (short)0, c10, false, false);
            c11 = __builtin_amdgcn_wmma_f32_16x16x32_bf16(false, a1, false, b1, (short)0, c11, false, false);
        }

        __syncthreads();      // all reads of buffer `cur` done before it is overwritten
        cur ^= 1u;
    }

    int m0 = m0blk + mw;
    int n0 = n0blk + nw;
    store_acc<MODE>(c00, m0,      n0,      lane, N, bias, a_logit, aux, of, ob);
    store_acc<MODE>(c01, m0,      n0 + 16, lane, N, bias, a_logit, aux, of, ob);
    store_acc<MODE>(c10, m0 + 16, n0,      lane, N, bias, a_logit, aux, of, ob);
    store_acc<MODE>(c11, m0 + 16, n0 + 16, lane, N, bias, a_logit, aux, of, ob);
}

// ---------------- sequential scan over S: h = a*h + sqrt(1-a^2)*xg ----------------
__global__ __launch_bounds__(256) void scan_kernel(const __bf16* __restrict__ a,
                                                   const __bf16* __restrict__ xg,
                                                   __bf16* __restrict__ h) {
    int t = blockIdx.x * blockDim.x + threadIdx.x;    // 0..B*H-1
    int b = t >> 10;
    int c = t & (Hc - 1);
    float hv = 0.f;
    size_t base = ((size_t)b * Sc) * Hc + c;
    for (int s = 0; s < Sc; ++s) {
        size_t idx = base + (size_t)s * Hc;
        float av = (float)a[idx];
        float xv = (float)xg[idx];
        hv = av * hv + sqrtf(fmaxf(0.f, 1.f - av * av)) * xv;
        h[idx] = (__bf16)hv;
    }
}

// ---------------- host ----------------
extern "C" void kernel_launch(void* const* d_in, const int* in_sizes, int n_in,
                              void* d_out, int out_size, void* d_ws, size_t ws_size,
                              hipStream_t stream) {
    (void)in_sizes; (void)n_in; (void)out_size; (void)ws_size;
    const float* x       = (const float*)d_in[0];
    const float* a_logit = (const float*)d_in[1];
    const float* W_dx = (const float*)d_in[2];  const float* b_dx = (const float*)d_in[3];
    const float* W_x  = (const float*)d_in[4];  const float* b_x  = (const float*)d_in[5];
    const float* W_gx = (const float*)d_in[6];  const float* b_gx = (const float*)d_in[7];
    const float* W_ga = (const float*)d_in[8];  const float* b_ga = (const float*)d_in[9];
    const float* W_o  = (const float*)d_in[10]; const float* b_o  = (const float*)d_in[11];
    float* out = (float*)d_out;

    // workspace carve-out (256B aligned chunks)
    char* ws = (char*)d_ws;
    size_t off = 0;
    auto carve = [&](size_t bytes) -> void* {
        void* p = ws + off;
        off += (bytes + 255) & ~(size_t)255;
        return p;
    };
    __bf16* Xf    = (__bf16*)carve((size_t)Mtot * KFP * sizeof(__bf16));
    __bf16* Wt_dx = (__bf16*)carve((size_t)DOUT * KFP * sizeof(__bf16));
    __bf16* Wt_x  = (__bf16*)carve((size_t)Hc   * KFP * sizeof(__bf16));
    __bf16* Wt_gx = (__bf16*)carve((size_t)Hc   * Hc  * sizeof(__bf16));
    __bf16* Wt_ga = (__bf16*)carve((size_t)Hc   * Hc  * sizeof(__bf16));
    __bf16* Wt_o  = (__bf16*)carve((size_t)DOUT * Hc  * sizeof(__bf16));
    __bf16* xh    = (__bf16*)carve((size_t)Mtot * Hc  * sizeof(__bf16));
    float*  dx    = (float*) carve((size_t)Mtot * DOUT * sizeof(float));
    __bf16* a_buf = (__bf16*)carve((size_t)Mtot * Hc  * sizeof(__bf16));
    __bf16* xg    = (__bf16*)carve((size_t)Mtot * Hc  * sizeof(__bf16));
    __bf16* h_buf = (__bf16*)carve((size_t)Mtot * Hc  * sizeof(__bf16));

    const int TB = 256;
    // prep feature matrix + weights
    int totX = Mtot * KFP;
    prep_x_kernel<<<(totX + TB - 1) / TB, TB, 0, stream>>>(x, Xf, totX);
    int totWp = DOUT * KFP;
    prep_wt_kernel<<<(totWp + TB - 1) / TB, TB, 0, stream>>>(W_dx, Wt_dx, KF, DOUT, KFP, totWp);
    prep_wt_kernel<<<(totWp + TB - 1) / TB, TB, 0, stream>>>(W_x,  Wt_x,  KF, Hc,   KFP, totWp);
    int totWs = Hc * Hc;
    prep_wt_kernel<<<(totWs + TB - 1) / TB, TB, 0, stream>>>(W_gx, Wt_gx, Hc, Hc, Hc, totWs);
    prep_wt_kernel<<<(totWs + TB - 1) / TB, TB, 0, stream>>>(W_ga, Wt_ga, Hc, Hc, Hc, totWs);
    prep_wt_kernel<<<(totWs + TB - 1) / TB, TB, 0, stream>>>(W_o,  Wt_o,  Hc, Hc, Hc, totWs);

    dim3 gridF(DOUT / 128, Mtot / 64);   // (8, 256)
    // dx = Xf @ W_dx + b_dx
    gemm_wmma_kernel<0><<<gridF, TB, 0, stream>>>(Xf, Wt_dx, KFP, DOUT, KFP, KFP,
                                                  b_dx, nullptr, nullptr, dx, nullptr);
    // xh = sqrt(1-sig(a)^2) * (Xf @ W_x + b_x)
    gemm_wmma_kernel<1><<<gridF, TB, 0, stream>>>(Xf, Wt_x, KFP, Hc, KFP, KFP,
                                                  b_x, a_logit, nullptr, nullptr, xh);
    dim3 gridS(Hc / 128, Mtot / 64);
    // xg = sigmoid(xh @ W_gx + b_gx) * xh
    gemm_wmma_kernel<2><<<gridS, TB, 0, stream>>>(xh, Wt_gx, Hc, Hc, Hc, Hc,
                                                  b_gx, nullptr, xh, nullptr, xg);
    // a = sigmoid(xh @ W_ga + b_ga) * sig(a_logit)
    gemm_wmma_kernel<3><<<gridS, TB, 0, stream>>>(xh, Wt_ga, Hc, Hc, Hc, Hc,
                                                  b_ga, a_logit, nullptr, nullptr, a_buf);
    // sequential scan -> h
    scan_kernel<<<(Bc * Hc) / TB, TB, 0, stream>>>(a_buf, xg, h_buf);
    // out = (dx + h @ W_o + b_o) / 2
    gemm_wmma_kernel<4><<<gridS, TB, 0, stream>>>(h_buf, Wt_o, Hc, DOUT, Hc, Hc,
                                                  b_o, nullptr, dx, out, nullptr);
}
